// WaveNet_962072674955
// MI455X (gfx1250) — matrix operand order repaired and checked by reading, not compiled, not run on hardware
//
#include <hip/hip_runtime.h>
#include <math.h>

typedef __bf16 bf16_t;
typedef __attribute__((ext_vector_type(8)))  __bf16 v8bf;
typedef __attribute__((ext_vector_type(16))) __bf16 v16bf;
typedef __attribute__((ext_vector_type(8)))  float  v8f;

#define T_LEN   4096
#define CCH     256
#define NLAYERS 40
#define BT      32      // time columns per block  -> 128 blocks (occupancy)
#define NT      2       // 16-wide N sub-tiles per block
#define THREADS 512     // 16 waves
#define ZSTR    264     // padded LDS column stride (bf16 elements)
#define TPAD    512     // zero prefix columns in hT buffers (max dilation)
#define ETT     64      // time columns per k_embed block

// ---------------------------------------------------------------------------
// WMMA helpers (CDNA5 wave32, 16x16x32 bf16 -> f32)
// ---------------------------------------------------------------------------
__device__ __forceinline__ v16bf cat8(v8bf lo, v8bf hi) {
  return __builtin_shufflevector(lo, hi, 0,1,2,3,4,5,6,7,8,9,10,11,12,13,14,15);
}

__device__ __forceinline__ v8f wmma_bf16(v16bf a, v16bf b, v8f c) {
  // 8 args: (neg_a, A, neg_b, B, c_mod, C, reuse_a, reuse_b)
  return __builtin_amdgcn_wmma_f32_16x16x32_bf16(false, a, false, b, (short)0, c,
                                                 false, false);
}

// A-matrix fragment (16x32 bf16, row-major): lanes 0-15 hold row m0+lane,
// K {k..k+15}; lanes 16-31 row m0+lane-16, K {k+16..k+31}.
__device__ __forceinline__ v16bf load_afrag(const bf16_t* __restrict__ rowp, int hiA) {
  v8bf lo = *(const v8bf*)(rowp + 2 * hiA);
  v8bf hi = *(const v8bf*)(rowp + 2 * hiA + 8);
  return cat8(lo, hi);
}

// B-matrix fragment (32x16 bf16): lane L holds 16 contiguous K values of column
// (L&15), starting at k0 + ((L>=16)?16:0).  p points at that K run.
__device__ __forceinline__ v16bf load_bfrag(const bf16_t* __restrict__ p) {
  v8bf lo = *(const v8bf*)(p);
  v8bf hi = *(const v8bf*)(p + 8);
  return cat8(lo, hi);
}

__device__ __forceinline__ unsigned pack2(float a, float b) {
  unsigned short ua = __builtin_bit_cast(unsigned short, (bf16_t)a);
  unsigned short ub = __builtin_bit_cast(unsigned short, (bf16_t)b);
  return (unsigned)ua | ((unsigned)ub << 16);
}

// ---------------------------------------------------------------------------
// Prolog kernels
// ---------------------------------------------------------------------------
// layer_w (NL,512,256,2) f32 -> wA[row][tap*256+c] bf16  (row = i*512+o)
__global__ void k_prep_layerw(const float* __restrict__ w, bf16_t* __restrict__ wA,
                              long n) {
  long idx = (long)blockIdx.x * blockDim.x + threadIdx.x;
  if (idx >= n) return;
  int  c   = (int)(idx & 255);
  int  tap = (int)((idx >> 8) & 1);
  long row = idx >> 9;
  wA[row * 512 + tap * 256 + c] = (bf16_t)w[(row * 256 + c) * 2 + tap];
}

__global__ void k_cvt(const float* __restrict__ src, bf16_t* __restrict__ dst, int n) {
  int i = blockIdx.x * blockDim.x + threadIdx.x;
  if (i < n) dst[i] = (bf16_t)src[i];
}

__global__ void k_zero(float* __restrict__ p, int n) {
  int i = blockIdx.x * blockDim.x + threadIdx.x;
  if (i < n) p[i] = 0.0f;
}

__global__ void k_zero_bf16(bf16_t* __restrict__ p, int n) {
  int i = blockIdx.x * blockDim.x + threadIdx.x;
  if (i < n) p[i] = (bf16_t)0.0f;
}

// h[c][t] = tanh(emb[x[t]][c]);  also transposed bf16 copy hT[t][c].
// LDS-staged so both global writes are coalesced.
__global__ __launch_bounds__(CCH)
void k_embed(const int* __restrict__ x, const float* __restrict__ emb,
             float* __restrict__ h, bf16_t* __restrict__ hT) {
  __shared__ float tile[ETT][CCH + 1];
  int t0 = blockIdx.x * ETT;
  int c  = threadIdx.x;
  for (int i = 0; i < ETT; ++i) {
    int cls = x[t0 + i];
    float v = tanhf(emb[(size_t)cls * CCH + c]);
    tile[i][c] = v;
    hT[(size_t)(t0 + i) * CCH + c] = (bf16_t)v;   // contiguous in c
  }
  __syncthreads();
  for (int idx = threadIdx.x; idx < ETT * CCH; idx += CCH) {
    int cc = idx >> 6;                             // 0..255
    int tl = idx & (ETT - 1);                      // 0..63
    h[(size_t)cc * T_LEN + t0 + tl] = tile[tl][cc];  // contiguous in tl
  }
}

// cond_small[g][f] = sum_a cond_w[g][a] * y[a][f]   (g < 20480, f < 16)
__global__ void k_cond(const float* __restrict__ cw, const float* __restrict__ y,
                       float* __restrict__ cs, int n) {
  int idx = blockIdx.x * blockDim.x + threadIdx.x;
  if (idx >= n) return;
  int g = idx >> 4;
  int f = idx & 15;
  float s = 0.0f;
  const float* row = cw + (size_t)g * 80;
#pragma unroll 8
  for (int a = 0; a < 80; ++a) s += row[a] * y[a * 16 + f];
  cs[idx] = s;
}

// ---------------------------------------------------------------------------
// Fused WaveNet layer:
//   z  = W0 @ h(t-d) + W1 @ h(t) + b + cond          (512 x BT per block)
//   za = tanh(z[:256]) * sigmoid(z[256:])            (bf16 tile in LDS)
//   o  = Wo @ za + bo ; h += o[:256] ; skip += o[256:]
// hT_in points past a TPAD-column zero prefix, so t-d indexing needs no branch.
// ---------------------------------------------------------------------------
__global__ __launch_bounds__(THREADS)
void k_layer(const bf16_t* __restrict__ hT_in, const float* __restrict__ h_in,
             float* __restrict__ h_out, bf16_t* __restrict__ hT_out,
             float* __restrict__ skip,
             const bf16_t* __restrict__ wA,   // [512][512] this layer (tap-major K)
             const float*  __restrict__ lb,   // [512]
             const bf16_t* __restrict__ woA,  // [512][256] (or [256][256] if last)
             const float*  __restrict__ wob,  // [512] (or [256] if last)
             const float*  __restrict__ cs,   // [512][16] cond for this layer
             int dil, int is_last) {
  __shared__ bf16_t lds_z[BT * ZSTR];

  const int lane = threadIdx.x & 31;
  const int wid  = threadIdx.x >> 5;   // 0..15
  const int ln   = lane & 15;
  const int hi16 = lane >> 4;          // 0 or 1
  const int hiA  = hi16 * 8;           // A fragment K offset
  const int hiB  = hi16 * 16;          // B fragment K offset
  const int hiC  = hi16 * 8;           // C/D row offset
  const int t0   = blockIdx.x * BT;

  // -------- conv GEMM: rows [mA, mA+16) = 'a' half, [mG, mG+16) = 'g' half ----
  const int mA = wid * 16;
  const int mG = 256 + wid * 16;
  v8f accA[NT], accG[NT];
#pragma unroll
  for (int nt = 0; nt < NT; ++nt) { accA[nt] = (v8f){}; accG[nt] = (v8f){}; }

  const bf16_t* arowA = wA + (size_t)(mA + ln) * 512;
  const bf16_t* arowG = wA + (size_t)(mG + ln) * 512;

  for (int tap = 0; tap < 2; ++tap) {
    // tap 0 reads h(t-d): negative t hits the zero prefix (no branch needed)
    const bf16_t* bbase =
        hT_in + (long)(t0 + ln - (tap == 0 ? dil : 0)) * CCH + hiB;
    for (int k0 = 0; k0 < 256; k0 += 32) {
      v16bf bfr[NT];
#pragma unroll
      for (int nt = 0; nt < NT; ++nt)
        bfr[nt] = load_bfrag(bbase + (long)nt * 16 * CCH + k0);
      v16bf afA = load_afrag(arowA + tap * 256 + k0, hiA);
      v16bf afG = load_afrag(arowG + tap * 256 + k0, hiA);
#pragma unroll
      for (int nt = 0; nt < NT; ++nt) {
        accA[nt] = wmma_bf16(afA, bfr[nt], accA[nt]);
        accG[nt] = wmma_bf16(afG, bfr[nt], accG[nt]);
      }
    }
  }

  // -------- bias + upsampled conditioning + gated activation -> LDS (bf16) ----
#pragma unroll
  for (int nt = 0; nt < NT; ++nt) {
    int   tl  = nt * 16 + ln;
    int   t   = t0 + tl;
    float pos = fmaxf((t + 0.5f) * (1.0f / 256.0f) - 0.5f, 0.0f);
    int   i0  = (int)pos;
    int   i1  = (i0 < 15) ? i0 + 1 : 15;
    float wfr = pos - (float)i0;
    float zv[8];
#pragma unroll
    for (int j = 0; j < 8; ++j) {
      int ra = mA + j + hiC;                 // 0..255
      int rg = mG + j + hiC;                 // 256..511
      const float* ca = cs + (size_t)ra * 16;
      const float* cg = cs + (size_t)rg * 16;
      float av = accA[nt][j] + lb[ra] + ca[i0] * (1.0f - wfr) + ca[i1] * wfr;
      float gv = accG[nt][j] + lb[rg] + cg[i0] * (1.0f - wfr) + cg[i1] * wfr;
      zv[j] = tanhf(av) * (1.0f / (1.0f + __expf(-gv)));
    }
    // rows (mA + 2j + hiC, +1) are adjacent bf16 in the [col][c] tile: pack pairs
#pragma unroll
    for (int j = 0; j < 8; j += 2) {
      int ra = mA + j + hiC;
      *(unsigned*)&lds_z[tl * ZSTR + ra] = pack2(zv[j], zv[j + 1]);
    }
  }
  __syncthreads();

  // -------- output projection GEMM --------------------------------------------
  if (!is_last) {
    const int mH = wid * 16;                 // residual rows
    const int mS = 256 + wid * 16;           // skip rows
    v8f accH[NT], accS[NT];
#pragma unroll
    for (int nt = 0; nt < NT; ++nt) { accH[nt] = (v8f){}; accS[nt] = (v8f){}; }
    const bf16_t* arH = woA + (size_t)(mH + ln) * 256;
    const bf16_t* arS = woA + (size_t)(mS + ln) * 256;
    for (int k0 = 0; k0 < 256; k0 += 32) {
      v16bf bfr[NT];
#pragma unroll
      for (int nt = 0; nt < NT; ++nt)
        bfr[nt] = load_bfrag(&lds_z[(nt * 16 + ln) * ZSTR + k0 + hiB]);
      v16bf aH = load_afrag(arH + k0, hiA);
      v16bf aS = load_afrag(arS + k0, hiA);
#pragma unroll
      for (int nt = 0; nt < NT; ++nt) {
        accH[nt] = wmma_bf16(aH, bfr[nt], accH[nt]);
        accS[nt] = wmma_bf16(aS, bfr[nt], accS[nt]);
      }
    }
#pragma unroll
    for (int nt = 0; nt < NT; ++nt) {
      int t = t0 + nt * 16 + ln;
      float hv[8];
#pragma unroll
      for (int j = 0; j < 8; ++j) {
        int rh = mH + j + hiC;
        int rs = mS + j + hiC;
        hv[j] = h_in[(size_t)rh * T_LEN + t] + accH[nt][j] + wob[rh];
        h_out[(size_t)rh * T_LEN + t] = hv[j];
        skip[(size_t)(rs - 256) * T_LEN + t] += accS[nt][j] + wob[rs];
      }
      // transposed bf16 copy: rows rh, rh+1 adjacent -> packed 32-bit stores
#pragma unroll
      for (int j = 0; j < 8; j += 2) {
        int rh = mH + j + hiC;
        *(unsigned*)&hT_out[(size_t)t * CCH + rh] = pack2(hv[j], hv[j + 1]);
      }
    }
  } else {
    const int mS = wid * 16;                 // last layer: skip only (256 rows)
    v8f accS[NT];
#pragma unroll
    for (int nt = 0; nt < NT; ++nt) accS[nt] = (v8f){};
    const bf16_t* arS = woA + (size_t)(mS + ln) * 256;
    for (int k0 = 0; k0 < 256; k0 += 32) {
      v16bf bfr[NT];
#pragma unroll
      for (int nt = 0; nt < NT; ++nt)
        bfr[nt] = load_bfrag(&lds_z[(nt * 16 + ln) * ZSTR + k0 + hiB]);
      v16bf aS = load_afrag(arS + k0, hiA);
#pragma unroll
      for (int nt = 0; nt < NT; ++nt) accS[nt] = wmma_bf16(aS, bfr[nt], accS[nt]);
    }
#pragma unroll
    for (int nt = 0; nt < NT; ++nt) {
      int t = t0 + nt * 16 + ln;
#pragma unroll
      for (int j = 0; j < 8; ++j) {
        int rs = mS + j + hiC;
        skip[(size_t)rs * T_LEN + t] += accS[nt][j] + wob[rs];
      }
    }
  }
}

// ---------------------------------------------------------------------------
// Epilog: out = end2 @ relu(end1 @ relu(skip) + b1) + b2
// ---------------------------------------------------------------------------
__global__ __launch_bounds__(THREADS)
void k_final(const float* __restrict__ skip,
             const bf16_t* __restrict__ e1, const float* __restrict__ e1b,
             const bf16_t* __restrict__ e2, const float* __restrict__ e2b,
             float* __restrict__ out) {
  __shared__ bf16_t ldsA[BT * ZSTR];
  __shared__ bf16_t ldsB[BT * ZSTR];

  const int lane = threadIdx.x & 31;
  const int wid  = threadIdx.x >> 5;
  const int ln   = lane & 15;
  const int hi16 = lane >> 4;
  const int hiA  = hi16 * 8;
  const int hiB  = hi16 * 16;
  const int hiC  = hi16 * 8;
  const int t0   = blockIdx.x * BT;

  // relu(skip) tile -> LDS, column-major bf16 (coalesced global reads)
  for (int idx = threadIdx.x; idx < BT * CCH; idx += THREADS) {
    int c   = idx >> 5;                        // 0..255
    int col = idx & (BT - 1);                  // 0..31
    float v = skip[(size_t)c * T_LEN + t0 + col];
    ldsA[col * ZSTR + c] = (bf16_t)fmaxf(v, 0.0f);
  }
  __syncthreads();

  const int m = wid * 16;                      // 16 waves x 16 rows = 256
  // GEMM 1: end1 @ relu(skip)
  v8f acc[NT];
#pragma unroll
  for (int nt = 0; nt < NT; ++nt) acc[nt] = (v8f){};
  const bf16_t* ar1 = e1 + (size_t)(m + ln) * 256;
  for (int k0 = 0; k0 < 256; k0 += 32) {
    v16bf bfr[NT];
#pragma unroll
    for (int nt = 0; nt < NT; ++nt)
      bfr[nt] = load_bfrag(&ldsA[(nt * 16 + ln) * ZSTR + k0 + hiB]);
    v16bf a1 = load_afrag(ar1 + k0, hiA);
#pragma unroll
    for (int nt = 0; nt < NT; ++nt) acc[nt] = wmma_bf16(a1, bfr[nt], acc[nt]);
  }
#pragma unroll
  for (int nt = 0; nt < NT; ++nt) {
    int tl = nt * 16 + ln;
    float rv[8];
#pragma unroll
    for (int j = 0; j < 8; ++j) {
      int r = m + j + hiC;
      rv[j] = fmaxf(acc[nt][j] + e1b[r], 0.0f);
    }
#pragma unroll
    for (int j = 0; j < 8; j += 2) {
      int r = m + j + hiC;
      *(unsigned*)&ldsB[tl * ZSTR + r] = pack2(rv[j], rv[j + 1]);
    }
  }
  __syncthreads();

  // GEMM 2: end2 @ relu(...)
#pragma unroll
  for (int nt = 0; nt < NT; ++nt) acc[nt] = (v8f){};
  const bf16_t* ar2 = e2 + (size_t)(m + ln) * 256;
  for (int k0 = 0; k0 < 256; k0 += 32) {
    v16bf bfr[NT];
#pragma unroll
    for (int nt = 0; nt < NT; ++nt)
      bfr[nt] = load_bfrag(&ldsB[(nt * 16 + ln) * ZSTR + k0 + hiB]);
    v16bf a2 = load_afrag(ar2 + k0, hiA);
#pragma unroll
    for (int nt = 0; nt < NT; ++nt) acc[nt] = wmma_bf16(a2, bfr[nt], acc[nt]);
  }
#pragma unroll
  for (int nt = 0; nt < NT; ++nt) {
    int t = t0 + nt * 16 + ln;
#pragma unroll
    for (int j = 0; j < 8; ++j) {
      int r = m + j + hiC;
      out[(size_t)r * T_LEN + t] = acc[nt][j] + e2b[r];
    }
  }
}

// ---------------------------------------------------------------------------
// Host launcher
// ---------------------------------------------------------------------------
extern "C" void kernel_launch(void* const* d_in, const int* in_sizes, int n_in,
                              void* d_out, int out_size, void* d_ws, size_t ws_size,
                              hipStream_t stream) {
  (void)in_sizes; (void)n_in; (void)out_size; (void)ws_size;
  const int*   x         = (const int*)  d_in[0];
  const float* y         = (const float*)d_in[1];
  /* d_in[2] = zeropad (unused by reference) */
  const float* emb_w     = (const float*)d_in[3];
  const float* cond_w    = (const float*)d_in[4];
  const float* layer_w   = (const float*)d_in[5];
  const float* layer_b   = (const float*)d_in[6];
  const float* wo_w      = (const float*)d_in[7];
  const float* wo_b      = (const float*)d_in[8];
  const float* last_wo_w = (const float*)d_in[9];
  const float* last_wo_b = (const float*)d_in[10];
  const float* end1_w    = (const float*)d_in[11];
  const float* end1_b    = (const float*)d_in[12];
  const float* end2_w    = (const float*)d_in[13];
  const float* end2_b    = (const float*)d_in[14];

  char*  ws  = (char*)d_ws;
  size_t off = 0;
  auto alloc = [&](size_t bytes) -> char* {
    char* p = ws + off;
    off += (bytes + 255) & ~(size_t)255;
    return p;
  };
  float*  hA       = (float*) alloc(sizeof(float)  * CCH * T_LEN);
  float*  hB       = (float*) alloc(sizeof(float)  * CCH * T_LEN);
  bf16_t* hTA_full = (bf16_t*)alloc(sizeof(bf16_t) * (size_t)(TPAD + T_LEN) * CCH);
  bf16_t* hTB_full = (bf16_t*)alloc(sizeof(bf16_t) * (size_t)(TPAD + T_LEN) * CCH);
  float*  skip     = (float*) alloc(sizeof(float)  * CCH * T_LEN);
  float*  cs       = (float*) alloc(sizeof(float)  * (size_t)NLAYERS * 512 * 16);
  bf16_t* wA       = (bf16_t*)alloc(sizeof(bf16_t) * (size_t)NLAYERS * 512 * 512);
  bf16_t* woA      = (bf16_t*)alloc(sizeof(bf16_t) * (size_t)(NLAYERS - 1) * 512 * 256);
  bf16_t* lwoA     = (bf16_t*)alloc(sizeof(bf16_t) * 256 * 256);
  bf16_t* e1A      = (bf16_t*)alloc(sizeof(bf16_t) * 256 * 256);
  bf16_t* e2A      = (bf16_t*)alloc(sizeof(bf16_t) * 256 * 256);
  bf16_t* hTA      = hTA_full + (size_t)TPAD * CCH;   // t index may go to -TPAD
  bf16_t* hTB      = hTB_full + (size_t)TPAD * CCH;

  // ---- weight conversion / rearrangement ----
  {
    long n = (long)NLAYERS * 512 * 512;
    k_prep_layerw<<<(unsigned)((n + 255) / 256), 256, 0, stream>>>(layer_w, wA, n);
  }
  {
    int n = (NLAYERS - 1) * 512 * 256;
    k_cvt<<<(n + 255) / 256, 256, 0, stream>>>(wo_w, woA, n);
  }
  k_cvt<<<(256 * 256 + 255) / 256, 256, 0, stream>>>(last_wo_w, lwoA, 256 * 256);
  k_cvt<<<(256 * 256 + 255) / 256, 256, 0, stream>>>(end1_w, e1A, 256 * 256);
  k_cvt<<<(256 * 256 + 255) / 256, 256, 0, stream>>>(end2_w, e2A, 256 * 256);

  // ---- zero causal pads, embedding, conditioning, skip init ----
  {
    int n = TPAD * CCH;
    k_zero_bf16<<<(n + 255) / 256, 256, 0, stream>>>(hTA_full, n);
    k_zero_bf16<<<(n + 255) / 256, 256, 0, stream>>>(hTB_full, n);
  }
  k_embed<<<T_LEN / ETT, CCH, 0, stream>>>(x, emb_w, hA, hTA);
  {
    int n = NLAYERS * 512 * 16;
    k_cond<<<(n + 255) / 256, 256, 0, stream>>>(cond_w, y, cs, n);
  }
  k_zero<<<(CCH * T_LEN + 255) / 256, 256, 0, stream>>>(skip, CCH * T_LEN);

  // ---- 40 dependent layer kernels (ping-pong h buffers) ----
  const float*  h_in   = hA;  float*  h_out  = hB;
  const bf16_t* hT_in  = hTA; bf16_t* hT_out = hTB;
  for (int i = 0; i < NLAYERS; ++i) {
    int dil  = 1 << (i % 10);
    int last = (i == NLAYERS - 1) ? 1 : 0;
    const bf16_t* wo_l  = last ? lwoA      : (woA + (size_t)i * 512 * 256);
    const float*  wob_l = last ? last_wo_b : (wo_b + (size_t)i * 512);
    k_layer<<<T_LEN / BT, THREADS, 0, stream>>>(
        hT_in, h_in, h_out, hT_out, skip,
        wA + (size_t)i * 512 * 512, layer_b + (size_t)i * 512,
        wo_l, wob_l, cs + (size_t)i * 512 * 16, dil, last);
    const float*  th = h_in;  h_in  = h_out;  h_out  = (float*)th;
    const bf16_t* tb = hT_in; hT_in = hT_out; hT_out = (bf16_t*)tb;
  }

  // ---- final two GEMMs -> logits ----
  k_final<<<T_LEN / BT, THREADS, 0, stream>>>(skip, e1A, end1_b, e2A, end2_b,
                                              (float*)d_out);
}
